// PointTransformerBlock_5609227289176
// MI455X (gfx1250) — matrix operand order repaired and checked by reading, not compiled, not run on hardware
//
#include <hip/hip_runtime.h>
#include <math.h>

// Problem constants (match reference)
#define BB   4
#define NPT  4096
#define CC   256
#define KNB  16
#define HH   8
#define DHD  32
#define PEH  64
#define FFH  1024
#define MTOT (BB * NPT)   // 16384 rows

typedef __bf16 bf16;
typedef __attribute__((ext_vector_type(16))) __bf16 v16bf;
typedef __attribute__((ext_vector_type(8)))  __bf16 v8bf;
typedef __attribute__((ext_vector_type(8)))  float  v8f;

__device__ __forceinline__ float gelu_f(float x) {
    return 0.5f * x * (1.0f + erff(x * 0.7071067811865475f));
}

// ---- WMMA fragment loaders (CDNA5 16-bit layouts, wave32) -------------------
// A: 16x32 (MxK), row-major in memory with leading dim lda (elements).
// Lane L<16: row M=L, K = 0..7 (vgpr0..3) and 16..23 (vgpr4..7)
// Lane L>=16: row M=L-16, K = 8..15 and 24..31
__device__ __forceinline__ v16bf load_a_frag(const bf16* base, int lda, int lane) {
    int row  = lane & 15;
    int koff = (lane >> 4) ? 8 : 0;
    const bf16* p = base + (size_t)row * lda + koff;
    v8bf lo = *(const v8bf*)(p);
    v8bf hi = *(const v8bf*)(p + 16);
    v16bf a;
#pragma unroll
    for (int i = 0; i < 8; ++i) { a[i] = lo[i]; a[i + 8] = hi[i]; }
    return a;
}

// B: 32x16 (KxN) fragment, sourced from transposed weights Bt[N][K] row-major.
// Lane L<16: col N=L, K = kk+0..7, kk+16..23 ; Lane L>=16: col N=L-16, K = kk+8..15, kk+24..31
__device__ __forceinline__ v16bf load_b_frag(const bf16* Bt, int ldb, int col0, int kk, int lane) {
    int col   = col0 + (lane & 15);
    int kbase = kk + ((lane >> 4) ? 8 : 0);
    const bf16* p = Bt + (size_t)col * ldb + kbase;
    v8bf lo = *(const v8bf*)(p);
    v8bf hi = *(const v8bf*)(p + 16);
    v16bf b;
#pragma unroll
    for (int i = 0; i < 8; ++i) { b[i] = lo[i]; b[i + 8] = hi[i]; }
    return b;
}

// ---- Weight prep: fp32 [K][N] -> bf16 transposed [N][K] ---------------------
__global__ __launch_bounds__(256)
void transpose_cvt_kernel(const float* __restrict__ in, bf16* __restrict__ out,
                          int Kd, int Nd) {
    int i = blockIdx.x * 256 + threadIdx.x;
    if (i >= Kd * Nd) return;
    int k = i / Nd, n = i % Nd;
    out[(size_t)n * Kd + k] = (bf16)in[i];
}

// ---- KNN: one thread per query, LDS-tiled candidates, register top-16 -------
__global__ __launch_bounds__(256)
void knn_kernel(const float* __restrict__ xyz, int* __restrict__ idx) {
    int blk = blockIdx.x;
    int b = blk >> 4;                 // 16 blocks per batch (4096/256)
    int qbase = (blk & 15) * 256;
    int tid = threadIdx.x;
    int qi = qbase + tid;
    const float* xb = xyz + (size_t)b * NPT * 3;
    float qx = xb[qi * 3 + 0], qy = xb[qi * 3 + 1], qz = xb[qi * 3 + 2];

    __shared__ float sx[256], sy[256], sz[256];
    float bd[KNB]; int bi[KNB];
#pragma unroll
    for (int j = 0; j < KNB; ++j) { bd[j] = 3.0e38f; bi[j] = 0; }

    for (int tb = 0; tb < NPT; tb += 256) {
        __syncthreads();
        sx[tid] = xb[(tb + tid) * 3 + 0];
        sy[tid] = xb[(tb + tid) * 3 + 1];
        sz[tid] = xb[(tb + tid) * 3 + 2];
        __syncthreads();
        for (int j = 0; j < 256; ++j) {
            int cand = tb + j;
            float dx = qx - sx[j], dy = qy - sy[j], dz = qz - sz[j];
            float dist = dx * dx + dy * dy + dz * dz;
            if (cand == qi) dist = 3.5e38f;     // exclude self
            if (dist < bd[KNB - 1]) {
                float cd = dist; int ci = cand;
#pragma unroll
                for (int t = 0; t < KNB; ++t) {
                    if (cd < bd[t]) {
                        float td = bd[t]; bd[t] = cd; cd = td;
                        int   ti = bi[t]; bi[t] = ci; ci = ti;
                    }
                }
            }
        }
    }
    int* op = idx + ((size_t)b * NPT + qi) * KNB;
#pragma unroll
    for (int j = 0; j < KNB; ++j) op[j] = bi[j];
}

// ---- LayerNorm (one wave32 per point, C=256 -> 8 vals/lane), bf16 out ------
__global__ __launch_bounds__(128)
void layernorm_kernel(const float* __restrict__ x, const float* __restrict__ g,
                      const float* __restrict__ beta, bf16* __restrict__ out,
                      int npts) {
    int wave = threadIdx.x >> 5, lane = threadIdx.x & 31;
    int p = blockIdx.x * 4 + wave;
    if (p >= npts) return;
    const float* xp = x + (size_t)p * CC;
    float4 a = ((const float4*)xp)[lane * 2 + 0];
    float4 c = ((const float4*)xp)[lane * 2 + 1];
    float s  = a.x + a.y + a.z + a.w + c.x + c.y + c.z + c.w;
    float ss = a.x*a.x + a.y*a.y + a.z*a.z + a.w*a.w
             + c.x*c.x + c.y*c.y + c.z*c.z + c.w*c.w;
#pragma unroll
    for (int o = 16; o > 0; o >>= 1) {
        s  += __shfl_xor(s,  o, 32);
        ss += __shfl_xor(ss, o, 32);
    }
    float mean = s * (1.0f / CC);
    float var  = ss * (1.0f / CC) - mean * mean;
    float inv  = rsqrtf(var + 1e-5f);
    int c0 = lane * 8;
    float vals[8] = { a.x, a.y, a.z, a.w, c.x, c.y, c.z, c.w };
    bf16* op = out + (size_t)p * CC;
#pragma unroll
    for (int i = 0; i < 8; ++i) {
        int cc = c0 + i;
        op[cc] = (bf16)((vals[i] - mean) * inv * g[cc] + beta[cc]);
    }
}

// ---- WMMA GEMM: one wave = 16x64 output (4 N-tiles), A-fragment reused ------
// A: bf16 [M][Kd] row-major. Bt: bf16 [N][Kd] row-major (transposed weights).
// All 4 B fragments are loaded before the 4 WMMAs so the matrix ops issue
// back-to-back while later loads are still in flight.
__global__ __launch_bounds__(32)
void gemm_wmma_kernel(const bf16* __restrict__ A, const bf16* __restrict__ Bt,
                      const float* __restrict__ bias, const float* __restrict__ resid,
                      float* __restrict__ outF, bf16* __restrict__ outB,
                      int Nd, int Kd, int fuse_gelu) {
    int nt0 = blockIdx.x * 4;       // first of 4 consecutive N-tiles
    int mt  = blockIdx.y;
    int lane = threadIdx.x;
    v8f acc[4] = {};
    const bf16* Arow = A + (size_t)mt * 16 * Kd;
    for (int kk = 0; kk < Kd; kk += 32) {
        // Prefetch the next K-slab of A (streaming activations; weights sit in L2)
        if (kk + 32 < Kd)
            __builtin_prefetch(Arow + (size_t)(lane & 15) * Kd + kk + 32, 0, 3);
        v16bf af = load_a_frag(Arow + kk, Kd, lane);   // reused across 4 B tiles
        v16bf bfr[4];
#pragma unroll
        for (int j = 0; j < 4; ++j)
            bfr[j] = load_b_frag(Bt, Kd, (nt0 + j) * 16, kk, lane);
#pragma unroll
        for (int j = 0; j < 4; ++j)
            acc[j] = __builtin_amdgcn_wmma_f32_16x16x32_bf16(
                false, af, false, bfr[j], (short)0, acc[j], false, false);
    }
    int rbase = (lane >> 4) ? 8 : 0;
#pragma unroll
    for (int j = 0; j < 4; ++j) {
        int col = (nt0 + j) * 16 + (lane & 15);
        float bv = bias ? bias[col] : 0.0f;
#pragma unroll
        for (int r = 0; r < 8; ++r) {
            int row = mt * 16 + rbase + r;
            size_t o = (size_t)row * Nd + col;
            float v = acc[j][r] + bv;
            if (fuse_gelu) v = gelu_f(v);
            if (resid) v += resid[o];
            if (outF) outF[o] = v;
            if (outB) outB[o] = (bf16)v;
        }
    }
}

// ---- Fused attention per point: PE-MLP (WMMA) + softmax over 16 neighbors ---
__global__ __launch_bounds__(128)
void attn_kernel(const float* __restrict__ xyz, const int* __restrict__ idx,
                 const float* __restrict__ q, const float* __restrict__ k,
                 const float* __restrict__ v,
                 const float* __restrict__ peW1, const float* __restrict__ peb1,
                 const bf16* __restrict__ peW2t, const float* __restrict__ peb2,
                 bf16* __restrict__ out) {
    int bn = blockIdx.x;            // 0 .. 16383
    int b  = bn >> 12;
    int n  = bn & (NPT - 1);
    int t = threadIdx.x, lane = t & 31, wave = t >> 5;

    __shared__ int   sidx[KNB];
    __shared__ float srel[KNB][3];
    __shared__ bf16  sA[KNB][PEH];          // gelu hidden, bf16 (WMMA A)
    __shared__ float spe[KNB][CC];          // pe vectors, f32
    __shared__ float slog[KNB][HH];
    __shared__ float sattn[KNB][HH];

    const float* xb = xyz + (size_t)b * NPT * 3;
    if (t < KNB) {
        int id = idx[(size_t)bn * KNB + t];
        sidx[t] = id;
        srel[t][0] = xb[n * 3 + 0] - xb[id * 3 + 0];
        srel[t][1] = xb[n * 3 + 1] - xb[id * 3 + 1];
        srel[t][2] = xb[n * 3 + 2] - xb[id * 3 + 2];
    }
    __syncthreads();

    // hidden = gelu(rel @ pe_W1 + pe_b1)   (16 x 64), tiny, VALU
#pragma unroll
    for (int e = t; e < KNB * PEH; e += 128) {
        int r = e >> 6, c = e & (PEH - 1);
        float val = srel[r][0] * peW1[c] + srel[r][1] * peW1[PEH + c]
                  + srel[r][2] * peW1[2 * PEH + c] + peb1[c];
        sA[r][c] = (bf16)gelu_f(val);
    }
    __syncthreads();

    // pe = hidden @ pe_W2 + pe_b2  via WMMA: (16x64)x(64x256).
    // Each wave owns 4 consecutive N-tiles; the two A fragments (K=0..31,
    // K=32..63) come from LDS once; both B fragments per tile are loaded
    // before the two back-to-back WMMAs.
    {
        v16bf a0 = load_a_frag(&sA[0][0] + 0,  PEH, lane);
        v16bf a1 = load_a_frag(&sA[0][0] + 32, PEH, lane);
        int rbase = (lane >> 4) ? 8 : 0;
#pragma unroll
        for (int j = 0; j < 4; ++j) {
            int nt = wave * 4 + j;
            v16bf b0 = load_b_frag(peW2t, PEH, nt * 16, 0, lane);
            v16bf b1 = load_b_frag(peW2t, PEH, nt * 16, 32, lane);
            v8f acc = {};
            acc = __builtin_amdgcn_wmma_f32_16x16x32_bf16(
                false, a0, false, b0, (short)0, acc, false, false);
            acc = __builtin_amdgcn_wmma_f32_16x16x32_bf16(
                false, a1, false, b1, (short)0, acc, false, false);
            int col = nt * 16 + (lane & 15);
            float bv = peb2[col];
#pragma unroll
            for (int r = 0; r < 8; ++r) spe[rbase + r][col] = acc[r] + bv;
        }
    }
    __syncthreads();

    // logits[nb][h] = q . (k_nbr + pe) / sqrt(DH)
    {
        int nb = t >> 3, h = t & 7;
        const float* qp = q + (size_t)bn * CC + h * DHD;
        const float* kp = k + ((size_t)(b * NPT + sidx[nb])) * CC + h * DHD;
        float acc = 0.0f;
#pragma unroll
        for (int d = 0; d < DHD; ++d)
            acc += qp[d] * (kp[d] + spe[nb][h * DHD + d]);
        slog[nb][h] = acc * 0.17677669529663687f;   // 1/sqrt(32)
    }
    __syncthreads();

    // softmax over neighbors (per head)
    if (t < HH) {
        float m = -3.0e38f;
#pragma unroll
        for (int nb = 0; nb < KNB; ++nb) m = fmaxf(m, slog[nb][t]);
        float s = 0.0f;
#pragma unroll
        for (int nb = 0; nb < KNB; ++nb) {
            float e = __expf(slog[nb][t] - m);
            sattn[nb][t] = e; s += e;
        }
        float inv = 1.0f / s;
#pragma unroll
        for (int nb = 0; nb < KNB; ++nb) sattn[nb][t] *= inv;
    }
    __syncthreads();

    // out[c] = sum_nb attn[nb][h(c)] * (v_nbr[c] + pe[nb][c]); 2 cols/thread
    bf16* op = out + (size_t)bn * CC;
#pragma unroll
    for (int cc = 0; cc < 2; ++cc) {
        int c = t * 2 + cc;
        int h = c >> 5;
        float acc = 0.0f;
#pragma unroll
        for (int nb = 0; nb < KNB; ++nb)
            acc += sattn[nb][h] * (v[((size_t)(b * NPT + sidx[nb])) * CC + c] + spe[nb][c]);
        op[c] = (bf16)acc;
    }
}

// ============================================================================
extern "C" void kernel_launch(void* const* d_in, const int* in_sizes, int n_in,
                              void* d_out, int out_size, void* d_ws, size_t ws_size,
                              hipStream_t stream) {
    const float* x      = (const float*)d_in[0];
    const float* xyz    = (const float*)d_in[1];
    /* d_in[2] = knn_chunk (unused) */
    const float* Wq     = (const float*)d_in[3];
    const float* Wk     = (const float*)d_in[4];
    const float* Wv     = (const float*)d_in[5];
    const float* peW1   = (const float*)d_in[6];
    const float* peb1   = (const float*)d_in[7];
    const float* peW2   = (const float*)d_in[8];
    const float* peb2   = (const float*)d_in[9];
    const float* projW  = (const float*)d_in[10];
    const float* projb  = (const float*)d_in[11];
    const float* ln1g   = (const float*)d_in[12];
    const float* ln1b   = (const float*)d_in[13];
    const float* ln2g   = (const float*)d_in[14];
    const float* ln2b   = (const float*)d_in[15];
    const float* ffW1   = (const float*)d_in[16];
    const float* ffb1   = (const float*)d_in[17];
    const float* ffW2   = (const float*)d_in[18];
    const float* ffb2   = (const float*)d_in[19];
    float* outp = (float*)d_out;

    // Workspace carve-up (256B aligned)
    char* ws = (char*)d_ws;
    size_t off = 0;
    auto alloc = [&](size_t bytes) -> char* {
        char* p = ws + off;
        off += (bytes + 255) & ~(size_t)255;
        return p;
    };
    int*  idx     = (int*) alloc((size_t)MTOT * KNB * 4);
    bf16* h1      = (bf16*)alloc((size_t)MTOT * CC * 2);
    float* qb     = (float*)alloc((size_t)MTOT * CC * 4);
    float* kb     = (float*)alloc((size_t)MTOT * CC * 4);
    float* vb     = (float*)alloc((size_t)MTOT * CC * 4);
    bf16* attn_o  = (bf16*)alloc((size_t)MTOT * CC * 2);
    float* x2     = (float*)alloc((size_t)MTOT * CC * 4);
    bf16* h2      = (bf16*)alloc((size_t)MTOT * CC * 2);
    bf16* u       = (bf16*)alloc((size_t)MTOT * FFH * 2);
    bf16* WqT     = (bf16*)alloc((size_t)CC * CC * 2);
    bf16* WkT     = (bf16*)alloc((size_t)CC * CC * 2);
    bf16* WvT     = (bf16*)alloc((size_t)CC * CC * 2);
    bf16* peW2T   = (bf16*)alloc((size_t)CC * PEH * 2);
    bf16* projWT  = (bf16*)alloc((size_t)CC * CC * 2);
    bf16* ffW1T   = (bf16*)alloc((size_t)FFH * CC * 2);
    bf16* ffW2T   = (bf16*)alloc((size_t)CC * FFH * 2);
    (void)ws_size; (void)in_sizes; (void)n_in; (void)out_size;

    auto tlaunch = [&](const float* in, bf16* out, int Kd, int Nd) {
        int tot = Kd * Nd;
        transpose_cvt_kernel<<<(tot + 255) / 256, 256, 0, stream>>>(in, out, Kd, Nd);
    };
    // Weight prep: fp32 [K][N] -> bf16 [N][K]
    tlaunch(Wq,    WqT,    CC,  CC);
    tlaunch(Wk,    WkT,    CC,  CC);
    tlaunch(Wv,    WvT,    CC,  CC);
    tlaunch(peW2,  peW2T,  PEH, CC);
    tlaunch(projW, projWT, CC,  CC);
    tlaunch(ffW1,  ffW1T,  CC,  FFH);
    tlaunch(ffW2,  ffW2T,  FFH, CC);

    // 1) KNN (top-16 neighbor indices)
    knn_kernel<<<BB * (NPT / 256), 256, 0, stream>>>(xyz, idx);

    // 2) LN1 -> h1 (bf16)
    layernorm_kernel<<<MTOT / 4, 128, 0, stream>>>(x, ln1g, ln1b, h1, MTOT);

    // 3) Q/K/V GEMMs (WMMA, no bias), f32 outputs for attention math
    dim3 gq(CC / 64, MTOT / 16);
    gemm_wmma_kernel<<<gq, 32, 0, stream>>>(h1, WqT, nullptr, nullptr, qb, nullptr, CC, CC, 0);
    gemm_wmma_kernel<<<gq, 32, 0, stream>>>(h1, WkT, nullptr, nullptr, kb, nullptr, CC, CC, 0);
    gemm_wmma_kernel<<<gq, 32, 0, stream>>>(h1, WvT, nullptr, nullptr, vb, nullptr, CC, CC, 0);

    // 4) Fused attention (PE MLP via WMMA + neighbor softmax), bf16 out
    attn_kernel<<<MTOT, 128, 0, stream>>>(xyz, idx, qb, kb, vb,
                                          peW1, peb1, peW2T, peb2, attn_o);

    // 5) Projection + residual: x2 = x + attn_o @ proj_W + proj_b
    gemm_wmma_kernel<<<gq, 32, 0, stream>>>(attn_o, projWT, projb, x, x2, nullptr, CC, CC, 0);

    // 6) LN2 -> h2 (bf16)
    layernorm_kernel<<<MTOT / 4, 128, 0, stream>>>(x2, ln2g, ln2b, h2, MTOT);

    // 7) FFN1: u = gelu(h2 @ ffn_W1 + b1) (bf16)
    dim3 gf1(FFH / 64, MTOT / 16);
    gemm_wmma_kernel<<<gf1, 32, 0, stream>>>(h2, ffW1T, ffb1, nullptr, nullptr, u, FFH, CC, 1);

    // 8) FFN2 + residual -> d_out: out = x2 + u @ ffn_W2 + b2
    gemm_wmma_kernel<<<gq, 32, 0, stream>>>(u, ffW2T, ffb2, x2, outp, nullptr, CC, FFH, 0);
}
